// PointAttentionNetwork_57724360458440
// MI455X (gfx1250) — matrix operand order repaired and checked by reading, not compiled, not run on hardware
//
#include <hip/hip_runtime.h>
#include <hip/hip_bf16.h>

// Problem constants (match reference)
constexpr int kB  = 4;
constexpr int kC  = 256;
constexpr int kN  = 4096;
constexpr int kCR = 64;
constexpr float kEPS = 1e-5f;

typedef __attribute__((ext_vector_type(16))) _Float16 v16h;
typedef __attribute__((ext_vector_type(8)))  _Float16 v8h;
typedef __attribute__((ext_vector_type(8)))  float    v8f;
typedef __attribute__((ext_vector_type(4)))  float    v4f;
typedef __attribute__((ext_vector_type(4)))  unsigned int u32x4;
typedef __attribute__((ext_vector_type(8)))  int          i32x8;
typedef __attribute__((ext_vector_type(4)))  int          i32x4;

// ---------------------------------------------------------------------------
// Fragment loaders for gfx1250 WMMA operands (CDNA5 ISA); A and B share contiguous k = 16h + e.
// ---------------------------------------------------------------------------

// B-matrix (16-bit, K=32 x N=16): lane holds col n = lane%16; element e holds
// K = (lane<16 ? 0 : 16) + e  -> 16 contiguous halves (works for global or LDS).
__device__ __forceinline__ v16h load_b_frag(const _Float16* p) {
  v8h lo = *(const v8h*)(p);
  v8h hi = *(const v8h*)(p + 8);
  v16h r;
#pragma unroll
  for (int e = 0; e < 8; ++e) { r[e] = lo[e]; r[e + 8] = hi[e]; }
  return r;
}

// A-matrix (16-bit, M=16 x K=32): lane holds row m = lane%16;
// kbase = (lane<16 ? 0 : 16); element e -> K=kbase+e
// -> 16 contiguous halves at p (caller pre-adds kbase).
__device__ __forceinline__ v16h load_a_frag(const _Float16* p) {
  v8h lo = *(const v8h*)(p);
  v8h hi = *(const v8h*)(p + 8);
  v16h r;
#pragma unroll
  for (int e = 0; e < 8; ++e) { r[e] = lo[e]; r[e + 8] = hi[e]; }
  return r;
}

// ---------------------------------------------------------------------------
// DPP16 row reductions: a softmax row lives in one 16-lane DPP row, so
// v_mov_dpp row_ror:{1,2,4,8} folds it with pure VALU (no ds_bpermute).
// ---------------------------------------------------------------------------
template <int CTRL>
__device__ __forceinline__ float dpp_ror(float x) {
  return __int_as_float(__builtin_amdgcn_update_dpp(
      0, __float_as_int(x), CTRL, 0xF, 0xF, true));
}
__device__ __forceinline__ float row_max16(float x) {
  x = fmaxf(x, dpp_ror<0x121>(x));  // row_ror:1
  x = fmaxf(x, dpp_ror<0x122>(x));  // row_ror:2
  x = fmaxf(x, dpp_ror<0x124>(x));  // row_ror:4
  x = fmaxf(x, dpp_ror<0x128>(x));  // row_ror:8
  return x;
}
__device__ __forceinline__ float row_sum16(float x) {
  x += dpp_ror<0x121>(x);
  x += dpp_ror<0x122>(x);
  x += dpp_ror<0x124>(x);
  x += dpp_ror<0x128>(x);
  return x;
}

// ---------------------------------------------------------------------------
// Tensor Data Mover: 2D tile load Global -> LDS with row padding
// (CDNA5 ISA D# layout; 6-arg clang-23 builtin form).
// data_size = 1 (2-byte elements); dims/strides in halves.
// ---------------------------------------------------------------------------
__device__ __forceinline__ void tdm_load_2d(
    unsigned lds_off, const _Float16* gptr,
    unsigned tensor_d0, unsigned tensor_d1,
    unsigned tile_d0, unsigned tile_d1,
    unsigned stride0, unsigned pad_interval, unsigned pad_amount) {
  const unsigned long long ga = (unsigned long long)(size_t)gptr;
  u32x4 g0;
  g0[0] = 1u;                                   // count=1, is_restore=0, no gather
  g0[1] = lds_off;                              // lds_addr[31:0]
  g0[2] = (unsigned)ga;                         // global_addr[31:0]
  g0[3] = (unsigned)(ga >> 32) | (2u << 30);    // global_addr[56:32] | type=2
  i32x8 g1;
  g1[0] = (int)((1u << 16) |                    // data_size = 2 bytes
                (1u << 20) |                    // pad_enable
                (pad_interval << 22) | (pad_amount << 25));
  g1[1] = (int)(tensor_d0 << 16);               // tensor_dim0[15:0] @ bits 63:48
  g1[2] = (int)((tensor_d0 >> 16) | (tensor_d1 << 16));
  g1[3] = (int)((tensor_d1 >> 16) | (tile_d0 << 16));
  g1[4] = (int)(tile_d1 & 0xFFFFu);             // tile_dim1; tile_dim2 = 0
  g1[5] = (int)stride0;                         // tensor_dim0_stride[31:0]
  g1[6] = 0;
  g1[7] = 0;
  const i32x4 z4 = {0, 0, 0, 0};
  const i32x8 z8 = {0, 0, 0, 0, 0, 0, 0, 0};
  __builtin_amdgcn_tensor_load_to_lds(g0, g1, z4, z4, z8, 0);
}

// ---------------------------------------------------------------------------
// Projection kernels: y = relu(scale*(W x) + shift), BN folded, f16 output.
// ---------------------------------------------------------------------------
__global__ __launch_bounds__(256) void proj_cr_kernel(
    const float* __restrict__ x, const float* __restrict__ w,
    const float* __restrict__ g, const float* __restrict__ bb,
    const float* __restrict__ mn, const float* __restrict__ vr,
    _Float16* __restrict__ ytr) {
  const int bo = blockIdx.x;
  const int b = bo / kCR, o = bo % kCR;
  const int n = blockIdx.y * 256 + threadIdx.x;
  const float* xp = x + (size_t)b * kC * kN + n;
  const float* wp = w + (size_t)o * kC;
  float acc = 0.f;
#pragma unroll 8
  for (int c = 0; c < kC; ++c) acc = fmaf(wp[c], xp[(size_t)c * kN], acc);
  const float sc = g[o] * rsqrtf(vr[o] + kEPS);
  const float sh = bb[o] - mn[o] * sc;
  const float y  = fmaxf(fmaf(acc, sc, sh), 0.f);
  ytr[((size_t)b * kN + n) * kCR + o] = (_Float16)y;
}

__global__ __launch_bounds__(256) void proj_c_kernel(
    const float* __restrict__ x, const float* __restrict__ w,
    const float* __restrict__ g, const float* __restrict__ bb,
    const float* __restrict__ mn, const float* __restrict__ vr,
    _Float16* __restrict__ y) {
  const int bo = blockIdx.x;
  const int b = bo / kC, o = bo % kC;
  const int n = blockIdx.y * 256 + threadIdx.x;
  const float* xp = x + (size_t)b * kC * kN + n;
  const float* wp = w + (size_t)o * kC;
  float acc = 0.f;
#pragma unroll 8
  for (int c = 0; c < kC; ++c) acc = fmaf(wp[c], xp[(size_t)c * kN], acc);
  const float sc = g[o] * rsqrtf(vr[o] + kEPS);
  const float sh = bb[o] - mn[o] * sc;
  const float yv = fmaxf(fmaf(acc, sc, sh), 0.f);
  y[((size_t)b * kC + o) * kN + n] = (_Float16)yv;
}

// ---------------------------------------------------------------------------
// Fused flash-attention kernel. One wave32 per 16-row Q tile; 4 waves/block
// cooperate on TDM-staged K/V tiles (double-buffered in LDS).
// Padded LDS row strides (halves): K rows 64->72, V rows 32->40, so the
// 16-lane b128 fragment reads hit 16 distinct bank groups.
// __launch_bounds__(128, 1): only ~1 wave/SIMD exists grid-wide, so give the
// scheduler the full VGPR budget to keep all fragments live (no dscnt-0
// serialization in the S phase).
// ---------------------------------------------------------------------------
constexpr int KROW = 72;   // 64 + 8  (pad_interval=4 -> 32 DW, pad_amount=3 -> 4 DW)
constexpr int VROW = 40;   // 32 + 8  (pad_interval=3 -> 16 DW, pad_amount=3 -> 4 DW)

__global__ __launch_bounds__(128, 1) void attn_kernel(
    const _Float16* __restrict__ Atr,   // [B, N, CR]
    const _Float16* __restrict__ Ktr,   // [B, N, CR]
    const _Float16* __restrict__ Dm,    // [B, C, N]
    const float*    __restrict__ label, // [B, C, N]
    float*          __restrict__ out)   // [B, C, N]
{
  __shared__ __align__(16) _Float16 lds_k[2][32 * KROW];   //  9.0 KB
  __shared__ __align__(16) _Float16 lds_v[2][256 * VROW];  // 40.0 KB
  __shared__ __align__(16) _Float16 lds_p[4][16 * 40];     //  5.0 KB

  const int wave  = threadIdx.x >> 5;
  const int lane  = threadIdx.x & 31;
  const int col   = lane & 15;
  const int lhalf = lane >> 4;
  const int b     = blockIdx.y;
  const int qbase = (blockIdx.x * 4 + wave) * 16;

  const _Float16* kglob = Ktr + (size_t)b * kN * kCR;
  const _Float16* vglob = Dm + (size_t)b * kC * kN;

  // Kick off TDM for the first K/V tiles (wave 0 drives the DMA).
  if (wave == 0) {
    tdm_load_2d((unsigned)(size_t)&lds_k[0][0], kglob,
                kCR, kN, kCR, 32, kCR, 4, 3);
    tdm_load_2d((unsigned)(size_t)&lds_v[0][0], vglob,
                kN, kC, 32, kC, kN, 3, 3);
  }

  // --- Q fragments (A-matrix role): K-dim = CR=64 -> 2 chunks of 32 ---
  const _Float16* qptr = Atr + ((size_t)b * kN + (qbase + col)) * kCR + lhalf * 16;
  const v16h qa0 = load_a_frag(qptr);        // c = 0..31
  const v16h qa1 = load_a_frag(qptr + 32);   // c = 32..63

  float mrow[8], lrow[8];
  v8f   Oacc[16];
#pragma unroll
  for (int i = 0; i < 8; ++i) { mrow[i] = -1e30f; lrow[i] = 0.f; }
#pragma unroll
  for (int t = 0; t < 16; ++t) Oacc[t] = v8f{};

  _Float16* const lp_wave = lds_p[wave];

  int buf = 0;
  for (int j0 = 0; j0 < kN; j0 += 32, buf ^= 1) {
    if (wave == 0) __builtin_amdgcn_s_wait_tensorcnt(0);
    __syncthreads();   // tile (K,V) for this step visible to all waves

    // Prefetch next tiles into the other buffer while we compute.
    if (wave == 0 && j0 + 32 < kN) {
      tdm_load_2d((unsigned)(size_t)&lds_k[buf ^ 1][0],
                  kglob + (size_t)(j0 + 32) * kCR,
                  kCR, kN, kCR, 32, kCR, 4, 3);
      tdm_load_2d((unsigned)(size_t)&lds_v[buf ^ 1][0],
                  vglob + (j0 + 32),
                  kN, kC, 32, kC, kN, 3, 3);
    }

    // ---- S = Q * K^T for 32 keys: two 16x16 tiles, each K=64 ----
    const _Float16* kb = &lds_k[buf][0];
    const v16h k00 = load_b_frag(kb + (size_t)col * KROW + lhalf * 16);
    const v16h k01 = load_b_frag(kb + (size_t)col * KROW + 32 + lhalf * 16);
    const v16h k10 = load_b_frag(kb + (size_t)(col + 16) * KROW + lhalf * 16);
    const v16h k11 = load_b_frag(kb + (size_t)(col + 16) * KROW + 32 + lhalf * 16);

    v8f s0 = v8f{}, s1 = v8f{};
    s0 = __builtin_amdgcn_wmma_f32_16x16x32_f16(false, qa0, false, k00, (short)0, s0, false, false);
    s1 = __builtin_amdgcn_wmma_f32_16x16x32_f16(false, qa0, false, k10, (short)0, s1, false, false);
    s0 = __builtin_amdgcn_wmma_f32_16x16x32_f16(false, qa1, false, k01, (short)0, s0, false, false);
    s1 = __builtin_amdgcn_wmma_f32_16x16x32_f16(false, qa1, false, k11, (short)0, s1, false, false);
    // Scheduling hint: issue all 8 ds_load_b128 first, then the 4 WMMAs,
    // so dscnt waits are partial instead of 0 before every WMMA.
    __builtin_amdgcn_sched_group_barrier(0x100, 8, 0);  // 8 DS reads
    __builtin_amdgcn_sched_group_barrier(0x008, 4, 0);  // 4 matrix ops

    // ---- Online softmax over the 32 new columns (DPP row reductions) ----
    float corr[8];
#pragma unroll
    for (int i = 0; i < 8; ++i) {
      const float tmax = row_max16(fmaxf(s0[i], s1[i]));
      const float mnew = fmaxf(mrow[i], tmax);
      const float p0 = __expf(s0[i] - mnew);
      const float p1 = __expf(s1[i] - mnew);
      const float psum = row_sum16(p0 + p1);
      const float c0 = __expf(mrow[i] - mnew);
      lrow[i] = lrow[i] * c0 + psum;
      mrow[i] = mnew;
      corr[i] = c0;
      _Float16* lp = lp_wave + (i + 8 * lhalf) * 40;
      lp[col]      = (_Float16)p0;
      lp[col + 16] = (_Float16)p1;
    }

#pragma unroll
    for (int t = 0; t < 16; ++t) {
#pragma unroll
      for (int i = 0; i < 8; ++i) Oacc[t][i] *= corr[i];
    }

    // ---- Reload P as an A-matrix fragment (per-wave LDS, in-order DS) ----
    const v16h pA = load_a_frag(lp_wave + (lane & 15) * 40 + lhalf * 16);

    // ---- O += P * V over 16 column tiles of C=256, V from LDS ----
    const _Float16* vb = &lds_v[buf][0];
    v16h vB = load_b_frag(vb + (size_t)col * VROW + lhalf * 16);
#pragma unroll
    for (int t = 0; t < 16; ++t) {
      v16h vNext = vB;
      if (t < 15)
        vNext = load_b_frag(vb + (size_t)((t + 1) * 16 + col) * VROW + lhalf * 16);
      Oacc[t] = __builtin_amdgcn_wmma_f32_16x16x32_f16(false, pA, false, vB, (short)0,
                                                       Oacc[t], false, false);
      vB = vNext;
    }
    // The next overwrite of lds_k/lds_v[buf] is only issued after the barrier
    // at the top of the *next* iteration, so no trailing barrier is needed.
  }

  // ---- Finalize: out[b,c,n] = label[b,c,n] + O[row, c] / l[row] ----
  float linv[8];
#pragma unroll
  for (int i = 0; i < 8; ++i) linv[i] = 1.0f / lrow[i];

  const size_t nb = (size_t)qbase + 8 * lhalf;  // rows i + 8*lhalf contiguous in N
#pragma unroll
  for (int t = 0; t < 16; ++t) {
    const int c = t * 16 + col;
    const size_t base = ((size_t)b * kC + c) * kN + nb;
    const v4f l0 = *(const v4f*)(label + base);
    const v4f l1 = *(const v4f*)(label + base + 4);
    v4f o0, o1;
#pragma unroll
    for (int i = 0; i < 4; ++i) {
      o0[i] = l0[i] + Oacc[t][i] * linv[i];
      o1[i] = l1[i] + Oacc[t][i + 4] * linv[i + 4];
    }
    *(v4f*)(out + base)     = o0;
    *(v4f*)(out + base + 4) = o1;
  }
}

// ---------------------------------------------------------------------------
extern "C" void kernel_launch(void* const* d_in, const int* in_sizes, int n_in,
                              void* d_out, int out_size, void* d_ws, size_t ws_size,
                              hipStream_t stream) {
  const float* label = (const float*)d_in[0];
  const float* ori   = (const float*)d_in[1];
  const float* w1 = (const float*)d_in[2];
  const float* g1 = (const float*)d_in[3];
  const float* b1 = (const float*)d_in[4];
  const float* m1 = (const float*)d_in[5];
  const float* v1 = (const float*)d_in[6];
  const float* w2 = (const float*)d_in[7];
  const float* g2 = (const float*)d_in[8];
  const float* b2 = (const float*)d_in[9];
  const float* m2 = (const float*)d_in[10];
  const float* v2 = (const float*)d_in[11];
  const float* w3 = (const float*)d_in[12];
  const float* g3 = (const float*)d_in[13];
  const float* b3 = (const float*)d_in[14];
  const float* m3 = (const float*)d_in[15];
  const float* v3 = (const float*)d_in[16];
  float* out = (float*)d_out;

  // Workspace: Atr (2 MB) | Ktr (2 MB) | D (8 MB), all f16.
  _Float16* Atr = (_Float16*)d_ws;
  _Float16* Ktr = Atr + (size_t)kB * kN * kCR;
  _Float16* Dm  = Ktr + (size_t)kB * kN * kCR;

  proj_cr_kernel<<<dim3(kB * kCR, kN / 256), dim3(256), 0, stream>>>(
      label, w1, g1, b1, m1, v1, Atr);
  proj_cr_kernel<<<dim3(kB * kCR, kN / 256), dim3(256), 0, stream>>>(
      ori, w2, g2, b2, m2, v2, Ktr);
  proj_c_kernel<<<dim3(kB * kC, kN / 256), dim3(256), 0, stream>>>(
      label, w3, g3, b3, m3, v3, Dm);

  attn_kernel<<<dim3(kN / 64, kB), dim3(128), 0, stream>>>(Atr, Ktr, Dm, label, out);
}
